// STCGNN_76424648065503
// MI455X (gfx1250) — compile-verified, matrix operand order checked by reading, not run on hardware
//
#include <hip/hip_runtime.h>

// ---------------- problem constants ----------------
constexpr int B = 64, T = 256, N = 83, F = 16;
constexpr int T1 = 254, T2 = 252;
constexpr int HID = 16, OUT = 64, NP = 96;   // nodes padded to 96
constexpr int TP = 260;                      // padded time extent (t reads reach 257)

// ---------------- workspace layout (bytes) ----------------
constexpr size_t OFF_L   = 0;                                  // 96*96 f16 Laplacian
constexpr size_t OFF_W1P = 18432;                              // tc1 packed: 3*2*32*16 f16
constexpr size_t OFF_W2P = OFF_W1P + 6144;                     // tc2 packed: 3*4*2*32*16 f16
constexpr size_t OFF_CWP = OFF_W2P + 24576;                    // cheb packed: 2*512 f16
constexpr size_t OFF_BN  = OFF_CWP + 2048;                     // bn sums [2*NP] f32
constexpr size_t OFF_SS  = OFF_BN + 1024;                      // scale/shift [2*NP] f32
constexpr size_t OFF_XH  = OFF_SS + 1024;                      // Xh: [B][N][TP][16] f16
constexpr size_t SZ_XH   = (size_t)B * N * TP * HID * 2;
constexpr size_t OFF_T0  = OFF_XH + SZ_XH;                     // T0: [B][T1][16][96] f16
constexpr size_t SZ_T0   = (size_t)B * T1 * HID * NP * 2;
constexpr size_t OFF_G   = OFF_XH;                             // G aliases Xh (Xh dead after k1)
constexpr size_t OFF_HL  = OFF_T0 + SZ_T0;                     // H_last: [B][N][64] f32

// ---------------- WMMA helpers ----------------
typedef __attribute__((ext_vector_type(16))) _Float16 v16h;
typedef __attribute__((ext_vector_type(8)))  _Float16 v8h;
typedef __attribute__((ext_vector_type(8)))  float    v8f;

__device__ __forceinline__ v8f wmma16(v16h a, v16h b, v8f c) {
    return __builtin_amdgcn_wmma_f32_16x16x32_f16(false, a, false, b,
                                                  (short)0, c, false, false);
}
__device__ __forceinline__ v16h cat8(v8h lo, v8h hi) {
    return __builtin_shufflevector(lo, hi, 0, 1, 2, 3, 4, 5, 6, 7,
                                   8, 9, 10, 11, 12, 13, 14, 15);
}
__device__ __forceinline__ int kmap(int lane, int h) {
    return ((h < 8) ? h : (h + 8)) + ((lane >> 4) << 3);
}
// fast GLU epilogue: relu(p * sigmoid(q) + r), v_rcp instead of IEEE divide
__device__ __forceinline__ float glu_relu(float p, float q, float r) {
    float s = __builtin_amdgcn_rcpf(1.f + __expf(-q));
    float hv = p * s + r;
    return hv > 0.f ? hv : 0.f;
}

// =========================================================
// k0_init: Laplacian (f16, 96x96, zero-padded) + zero BN accumulators
// =========================================================
__global__ __launch_bounds__(128) void k0_init(const int* __restrict__ ei, int nedges,
                                               _Float16* __restrict__ Lg,
                                               float* __restrict__ bnsum) {
    __shared__ float deg[NP];
    int tid = threadIdx.x;
    for (int i = tid; i < NP; i += 128) deg[i] = 0.f;
    for (int i = tid; i < 2 * NP; i += 128) bnsum[i] = 0.f;
    for (int i = tid; i < NP * NP; i += 128) Lg[i] = (_Float16)0.f;
    __syncthreads();
    for (int e = tid; e < nedges; e += 128) atomicAdd(&deg[ei[e]], 1.f);
    __syncthreads();
    for (int i = tid; i < NP; i += 128) deg[i] = (deg[i] > 0.f) ? rsqrtf(deg[i]) : 0.f;
    __syncthreads();
    for (int e = tid; e < nedges; e += 128) {
        int r = ei[e], c = ei[e + nedges];
        Lg[c * NP + r] = (_Float16)(-(deg[r] * deg[c]));
    }
}

// =========================================================
// k0_pack: pre-pack all weights into WMMA-ready f16 operand layouts
// =========================================================
__global__ __launch_bounds__(256) void k0_pack(
    const float* __restrict__ w11, const float* __restrict__ w12, const float* __restrict__ w13,
    const float* __restrict__ w21, const float* __restrict__ w22, const float* __restrict__ w23,
    const float* __restrict__ chebw,
    _Float16* __restrict__ W1P, _Float16* __restrict__ W2P, _Float16* __restrict__ CWP) {
    const int tid = threadIdx.x;
    for (int idx = tid; idx < 3 * 2 * 32 * 16; idx += 256) {
        int h = idx & 15, lane = (idx >> 4) & 31, chunk = (idx >> 9) & 1, j = idx >> 10;
        const float* w = (j == 0) ? w11 : (j == 1) ? w12 : w13;
        int o = lane & 15;
        int K = kmap(lane, h) + chunk * 32;
        W1P[idx] = (K < 48) ? (_Float16)w[o * 48 + (K & 15) * 3 + (K >> 4)] : (_Float16)0.f;
    }
    for (int idx = tid; idx < 3 * 4 * 2 * 32 * 16; idx += 256) {
        int h = idx & 15, lane = (idx >> 4) & 31, chunk = (idx >> 9) & 1;
        int ot = (idx >> 10) & 3, j = idx >> 12;
        const float* w = (j == 0) ? w21 : (j == 1) ? w22 : w23;
        int o = ot * 16 + (lane & 15);
        int K = kmap(lane, h) + chunk * 32;
        W2P[idx] = (K < 48) ? (_Float16)w[o * 48 + (K & 15) * 3 + (K >> 4)] : (_Float16)0.f;
    }
    for (int idx = tid; idx < 1024; idx += 256) {
        int h = idx & 15, lane = (idx >> 4) & 31, tab = idx >> 9;
        int nn = lane & 15;
        int K = kmap(lane, h);
        float v;
        if (tab == 0) v = (K < 16) ? chebw[K * 16 + nn] : chebw[256 + (K - 16) * 16 + nn];
        else          v = (K < 16) ? chebw[512 + K * 16 + nn] : 0.f;
        CWP[idx] = (_Float16)v;
    }
}

// =========================================================
// kX: stage X -> f16, transposed to [b][n][TP][16] with zero time-padding
// =========================================================
__global__ __launch_bounds__(256) void kX_stage(const float* __restrict__ X,
                                                _Float16* __restrict__ Xh) {
    const int n = blockIdx.x, b = blockIdx.y;
    const size_t dst = ((size_t)b * N + n) * TP * HID;
    for (int idx = threadIdx.x; idx < TP * HID; idx += 256) {
        int t = idx >> 4, c = idx & 15;
        float v = (t < T) ? X[(((size_t)b * T + t) * N + n) * F + c] : 0.f;
        Xh[dst + idx] = (_Float16)v;
    }
}

// =========================================================
// k1: TemporalConv1 (16 -> 16) + GLU via WMMA. One wave per (b, n, 16 t-steps).
// Writes T0 channel-major [b][t][16][96]; waves with n in [83,96) write zeros.
// =========================================================
__global__ __launch_bounds__(32) void k1_tconv1(
    const _Float16* __restrict__ Xh, const _Float16* __restrict__ W1P,
    const float* __restrict__ b1, const float* __restrict__ b2, const float* __restrict__ b3,
    _Float16* __restrict__ T0out) {
    const int lane = threadIdx.x;
    const int t0 = blockIdx.x * 16;
    const int n = blockIdx.y;          // 0..95
    const int b = blockIdx.z;
    const int o = lane & 15;
    const int hi = (lane >> 4) << 3;   // 0 or 8
    const int t0c = t0 + hi;
    _Float16* outp = T0out + (((size_t)b * T1 + t0c) * HID + o) * NP + n;
    constexpr size_t OSTR = (size_t)HID * NP;   // halves per t step

    if (n >= N) {   // zero-fill node padding (uniform per wave)
        if (t0 + 16 <= T1) {
#pragma unroll
            for (int r = 0; r < 8; ++r) outp[r * OSTR] = (_Float16)0.f;
        } else {
#pragma unroll
            for (int r = 0; r < 8; ++r)
                if (t0c + r < T1) outp[r * OSTR] = (_Float16)0.f;
        }
        return;
    }

    const int m = lane & 15;
    const _Float16* Xbn = Xh + ((size_t)b * N + n) * TP * HID;
    v8h a0lo = *(const v8h*)(Xbn + (size_t)(t0 + m) * HID + hi);
    v8h a0hi = *(const v8h*)(Xbn + (size_t)(t0 + m + 1) * HID + hi);
    v8h a1lo = *(const v8h*)(Xbn + (size_t)(t0 + m + 2) * HID + hi);
    v8h zz = (v8h)(_Float16)0.f;
    v16h A0 = cat8(a0lo, a0hi);
    v16h A1 = cat8(a1lo, zz);

    auto loadB = [&](int j, int chunk) -> v16h {
        return *(const v16h*)(W1P + (((j * 2 + chunk) * 32) + lane) * 16);
    };
    v8f accP = {}, accQ = {}, accR = {};
    accP = wmma16(A0, loadB(0, 0), accP); accP = wmma16(A1, loadB(0, 1), accP);
    accQ = wmma16(A0, loadB(1, 0), accQ); accQ = wmma16(A1, loadB(1, 1), accQ);
    accR = wmma16(A0, loadB(2, 0), accR); accR = wmma16(A1, loadB(2, 1), accR);

    const float bp = b1[o], bq = b2[o], br = b3[o];
    if (t0 + 16 <= T1) {   // full tile: unconditional stores, immediate offsets
#pragma unroll
        for (int r = 0; r < 8; ++r)
            outp[r * OSTR] = (_Float16)glu_relu(accP[r] + bp, accQ[r] + bq, accR[r] + br);
    } else {
#pragma unroll
        for (int r = 0; r < 8; ++r)
            if (t0c + r < T1)
                outp[r * OSTR] = (_Float16)glu_relu(accP[r] + bp, accQ[r] + bq, accR[r] + br);
    }
}

// =========================================================
// k2: ChebConv K=3. 4 waves/block share L in LDS; each wave owns one (b,t).
// Activation tiles are channel-major [16][96] in LDS -> contiguous B-operands
// and contiguous (v8h) C/D write-back.
// =========================================================
__global__ __launch_bounds__(128) void k2_cheb(
    const _Float16* __restrict__ Lg, const _Float16* __restrict__ T0,
    const _Float16* __restrict__ CWP, const float* __restrict__ chebb,
    _Float16* __restrict__ G) {
    __shared__ __align__(32) _Float16 Lsh[NP * NP];
    __shared__ __align__(32) _Float16 T0ch[4][HID * NP];
    __shared__ __align__(32) _Float16 Tx1ch[4][HID * NP];
    __shared__ __align__(32) _Float16 Tx2ch[4][HID * NP];

    const int tid = threadIdx.x, lane = tid & 31, w = tid >> 5;

    {   // cooperative load of L (18432 B)
        const uint4* src = (const uint4*)Lg;
        uint4* dst = (uint4*)Lsh;
        for (int i = tid; i < NP * NP / 8; i += 128) dst[i] = src[i];
    }
    __syncthreads();

    const int site = blockIdx.x * 4 + w;
    const int b = site / T1;
    const int t = site - b * T1;
    _Float16* t0s  = T0ch[w];
    _Float16* tx1s = Tx1ch[w];
    _Float16* tx2s = Tx2ch[w];

    {   // load T0[b][t] (16x96 f16, contiguous 3072 B)
        const uint4* src = (const uint4*)(T0 + ((size_t)b * T1 + t) * HID * NP);
        uint4* dst = (uint4*)t0s;
#pragma unroll
        for (int i = 0; i < 6; ++i) dst[lane + 32 * i] = src[lane + 32 * i];
    }
    __builtin_amdgcn_wave_barrier();

    const int nn = lane & 15;
    const int cb = (lane >> 4) << 3;   // 0 or 8 (K-offset, also C/D row offset)
    const v16h cwA = *(const v16h*)(CWP + lane * 16);
    const v16h cwB = *(const v16h*)(CWP + 512 + lane * 16);

    // ---- Tx1 = L @ T0 ----
#pragma unroll
    for (int mt = 0; mt < 6; ++mt) {
        v8f acc = {};
        const int row = mt * 16 + nn;
#pragma unroll
        for (int kc = 0; kc < 3; ++kc) {
            v16h a = cat8(*(const v8h*)(Lsh + row * NP + kc * 32 + cb),
                          *(const v8h*)(Lsh + row * NP + kc * 32 + cb + 16));
            v16h bm = cat8(*(const v8h*)(t0s + nn * NP + kc * 32 + cb),
                           *(const v8h*)(t0s + nn * NP + kc * 32 + cb + 16));
            acc = wmma16(a, bm, acc);
        }
        v8h pk;
#pragma unroll
        for (int r = 0; r < 8; ++r) pk[r] = (_Float16)acc[r];
        *(v8h*)(tx1s + nn * NP + mt * 16 + cb) = pk;   // rows mt*16+cb..+7 contiguous
    }
    __builtin_amdgcn_wave_barrier();

    // ---- Tx2 = 2 * L @ Tx1 - T0 ----
#pragma unroll
    for (int mt = 0; mt < 6; ++mt) {
        v8f acc = {};
        const int row = mt * 16 + nn;
#pragma unroll
        for (int kc = 0; kc < 3; ++kc) {
            v16h a = cat8(*(const v8h*)(Lsh + row * NP + kc * 32 + cb),
                          *(const v8h*)(Lsh + row * NP + kc * 32 + cb + 16));
            v16h bm = cat8(*(const v8h*)(tx1s + nn * NP + kc * 32 + cb),
                           *(const v8h*)(tx1s + nn * NP + kc * 32 + cb + 16));
            acc = wmma16(a, bm, acc);
        }
        v8h t0v = *(const v8h*)(t0s + nn * NP + mt * 16 + cb);
        v8h pk;
#pragma unroll
        for (int r = 0; r < 8; ++r)
            pk[r] = (_Float16)(2.f * acc[r] - (float)t0v[r]);
        *(v8h*)(tx2s + nn * NP + mt * 16 + cb) = pk;
    }
    __builtin_amdgcn_wave_barrier();

    // ---- G = relu([T0|Tx1]*[W0;W1] + [Tx2|0]*[W2;0] + bias) ----
    const float bias = chebb[nn];
    constexpr size_t GSTR = (size_t)TP * HID;   // halves per node row
#pragma unroll
    for (int mt = 0; mt < 6; ++mt) {
        const int Mg = mt * 16 + nn;
        v16h a1, a2;
#pragma unroll
        for (int h = 0; h < 16; ++h) {
            int K = kmap(lane, h);
            if (K < 16) {
                a1[h] = t0s[K * NP + Mg];
                a2[h] = tx2s[K * NP + Mg];
            } else {
                a1[h] = tx1s[(K - 16) * NP + Mg];
                a2[h] = (_Float16)0.f;
            }
        }
        v8f acc = {};
        acc = wmma16(a1, cwA, acc);
        acc = wmma16(a2, cwB, acc);

        _Float16* gp = G + (((size_t)b * N + mt * 16 + cb) * TP + t) * HID + nn;
        if (mt * 16 + 16 <= N) {   // folded per unrolled mt: mt 0..4 full
#pragma unroll
            for (int r = 0; r < 8; ++r) {
                float g = acc[r] + bias;
                gp[r * GSTR] = (_Float16)(g > 0.f ? g : 0.f);
            }
        } else {
#pragma unroll
            for (int r = 0; r < 8; ++r) {
                if (mt * 16 + cb + r < N) {
                    float g = acc[r] + bias;
                    gp[r * GSTR] = (_Float16)(g > 0.f ? g : 0.f);
                }
            }
        }
    }
}

// =========================================================
// k3: TemporalConv2 (16 -> 64) + GLU; per-node BN sum/sumsq + last-timestep store.
// =========================================================
__global__ __launch_bounds__(32) void k3_tconv2(
    const _Float16* __restrict__ G, const _Float16* __restrict__ W2P,
    const float* __restrict__ b1, const float* __restrict__ b2, const float* __restrict__ b3,
    float* __restrict__ Hlast, float* __restrict__ bnsum) {
    const int lane = threadIdx.x;
    const int t0 = blockIdx.x * 16;
    const int n = blockIdx.y;          // 0..82
    const int b = blockIdx.z;
    const int m = lane & 15;
    const int hi = (lane >> 4) << 3;
    const int t0c = t0 + hi;
    const _Float16* Gbn = G + ((size_t)b * N + n) * TP * HID;

    v8h a0lo = *(const v8h*)(Gbn + (size_t)(t0 + m) * HID + hi);
    v8h a0hi = *(const v8h*)(Gbn + (size_t)(t0 + m + 1) * HID + hi);
    v8h a1lo = *(const v8h*)(Gbn + (size_t)(t0 + m + 2) * HID + hi);
    v8h zz = (v8h)(_Float16)0.f;
    v16h A0 = cat8(a0lo, a0hi);
    v16h A1 = cat8(a1lo, zz);

    const bool full = (t0 + 16 <= T2);   // uniform
    float s1 = 0.f, s2 = 0.f;
#pragma unroll
    for (int ot = 0; ot < 4; ++ot) {
        const int o = ot * 16 + (lane & 15);
        auto loadB = [&](int j, int chunk) -> v16h {
            return *(const v16h*)(W2P + ((((j * 4 + ot) * 2 + chunk) * 32) + lane) * 16);
        };
        v8f accP = {}, accQ = {}, accR = {};
        accP = wmma16(A0, loadB(0, 0), accP); accP = wmma16(A1, loadB(0, 1), accP);
        accQ = wmma16(A0, loadB(1, 0), accQ); accQ = wmma16(A1, loadB(1, 1), accQ);
        accR = wmma16(A0, loadB(2, 0), accR); accR = wmma16(A1, loadB(2, 1), accR);
        const float bp = b1[o], bq = b2[o], br = b3[o];
        if (full) {
#pragma unroll
            for (int r = 0; r < 8; ++r) {
                float hv = glu_relu(accP[r] + bp, accQ[r] + bq, accR[r] + br);
                s1 += hv;
                s2 += hv * hv;
            }
        } else {
#pragma unroll
            for (int r = 0; r < 8; ++r) {
                if (t0c + r < T2) {
                    float hv = glu_relu(accP[r] + bp, accQ[r] + bq, accR[r] + br);
                    s1 += hv;
                    s2 += hv * hv;
                    if (t0c + r == T2 - 1)
                        Hlast[((size_t)b * N + n) * OUT + o] = hv;
                }
            }
        }
    }
#pragma unroll
    for (int off = 16; off > 0; off >>= 1) {
        s1 += __shfl_xor(s1, off, 32);
        s2 += __shfl_xor(s2, off, 32);
    }
    if (lane == 0) {
        atomicAdd(&bnsum[n], s1);
        atomicAdd(&bnsum[NP + n], s2);
    }
}

// =========================================================
// k35: BatchNorm affine params per node (training-mode batch stats)
// =========================================================
__global__ __launch_bounds__(NP) void k35_bn(const float* __restrict__ bnsum,
                                             const float* __restrict__ gamma,
                                             const float* __restrict__ beta,
                                             float* __restrict__ ss) {
    int n = threadIdx.x;
    if (n < N) {
        const float cnt = (float)((size_t)B * T2 * OUT);
        float mu = bnsum[n] / cnt;
        float var = bnsum[NP + n] / cnt - mu * mu;
        float sc = gamma[n] * rsqrtf(var + 1e-5f);
        ss[n] = sc;
        ss[NP + n] = beta[n] - mu * sc;
    }
}

// =========================================================
// k4: relu(BN(H_last)) -> mean over nodes -> Linear(64->12) -> concat extras
// =========================================================
__global__ __launch_bounds__(64) void k4_final(const float* __restrict__ Hlast,
                                               const float* __restrict__ ss,
                                               const float* __restrict__ hideout,
                                               const float* __restrict__ tstep,
                                               const float* __restrict__ linw,
                                               const float* __restrict__ linb,
                                               float* __restrict__ out) {
    __shared__ float sv[OUT];
    const int b = blockIdx.x, k = threadIdx.x;
    float s = 0.f;
    for (int n = 0; n < N; ++n) {
        float v = Hlast[((size_t)b * N + n) * OUT + k] * ss[n] + ss[NP + n];
        s += v > 0.f ? v : 0.f;
    }
    sv[k] = s * (1.f / (float)N);
    __syncthreads();
    if (k < 12) {
        float o = linb[k];
#pragma unroll
        for (int j = 0; j < OUT; ++j) o += sv[j] * linw[k * OUT + j];
        out[b * 15 + k] = o;
    } else if (k == 12) {
        out[b * 15 + 12] = hideout[b * 2];
    } else if (k == 13) {
        out[b * 15 + 13] = hideout[b * 2 + 1];
    } else if (k == 14) {
        out[b * 15 + 14] = tstep[b];
    }
}

// =========================================================
extern "C" void kernel_launch(void* const* d_in, const int* in_sizes, int n_in,
                              void* d_out, int out_size, void* d_ws, size_t ws_size,
                              hipStream_t stream) {
    const float* X       = (const float*)d_in[0];
    const float* hideout = (const float*)d_in[1];
    const float* tstep   = (const float*)d_in[2];
    const int*   ei      = (const int*)d_in[4];
    const float* tc1w1 = (const float*)d_in[5];  const float* tc1b1 = (const float*)d_in[6];
    const float* tc1w2 = (const float*)d_in[7];  const float* tc1b2 = (const float*)d_in[8];
    const float* tc1w3 = (const float*)d_in[9];  const float* tc1b3 = (const float*)d_in[10];
    const float* chebw = (const float*)d_in[11]; const float* chebb = (const float*)d_in[12];
    const float* tc2w1 = (const float*)d_in[13]; const float* tc2b1 = (const float*)d_in[14];
    const float* tc2w2 = (const float*)d_in[15]; const float* tc2b2 = (const float*)d_in[16];
    const float* tc2w3 = (const float*)d_in[17]; const float* tc2b3 = (const float*)d_in[18];
    const float* gamma = (const float*)d_in[19]; const float* beta  = (const float*)d_in[20];
    const float* linw  = (const float*)d_in[21]; const float* linb  = (const float*)d_in[22];

    char* ws = (char*)d_ws;
    _Float16* Lg    = (_Float16*)(ws + OFF_L);
    _Float16* W1P   = (_Float16*)(ws + OFF_W1P);
    _Float16* W2P   = (_Float16*)(ws + OFF_W2P);
    _Float16* CWP   = (_Float16*)(ws + OFF_CWP);
    float*    bnsum = (float*)(ws + OFF_BN);
    float*    ssb   = (float*)(ws + OFF_SS);
    _Float16* Xh    = (_Float16*)(ws + OFF_XH);
    _Float16* T0b   = (_Float16*)(ws + OFF_T0);
    _Float16* Gb    = (_Float16*)(ws + OFF_G);
    float*    Hlast = (float*)(ws + OFF_HL);
    const int nedges = in_sizes[4] / 2;

    k0_init<<<1, 128, 0, stream>>>(ei, nedges, Lg, bnsum);
    k0_pack<<<1, 256, 0, stream>>>(tc1w1, tc1w2, tc1w3, tc2w1, tc2w2, tc2w3, chebw,
                                   W1P, W2P, CWP);
    kX_stage<<<dim3(N, B), 256, 0, stream>>>(X, Xh);

    k1_tconv1<<<dim3(16, NP, B), 32, 0, stream>>>(Xh, W1P, tc1b1, tc1b2, tc1b3, T0b);

    k2_cheb<<<(B * T1) / 4, 128, 0, stream>>>(Lg, T0b, CWP, chebb, Gb);

    k3_tconv2<<<dim3(16, N, B), 32, 0, stream>>>(Gb, W2P, tc2b1, tc2b2, tc2b3,
                                                 Hlast, bnsum);

    k35_bn<<<1, NP, 0, stream>>>(bnsum, gamma, beta, ssb);

    k4_final<<<B, 64, 0, stream>>>(Hlast, ssb, hideout, tstep, linw, linb, (float*)d_out);
}
